// NeuralMachineTranslation_62328565399982
// MI455X (gfx1250) — matrix operand order repaired and checked by reading, not compiled, not run on hardware
//
#include <hip/hip_runtime.h>
#include <math.h>

// ---------------------------------------------------------------------------
// NMT seq2seq forward for MI455X (gfx1250), wave32 + v_wmma_f32_16x16x32_f16.
// V=50258, E=H=256, B=32, S=96, T=64.
// ---------------------------------------------------------------------------

#define VOCAB 50258
#define HID   256
#define G4H   1024      // 4*H
#define BATCH 32
#define SRCL  96
#define TRGL  64
#define DSTEP 63        // T-1 decoder steps

typedef __attribute__((ext_vector_type(16))) _Float16 v16h;
typedef __attribute__((ext_vector_type(8)))  float    v8f;

// ---------------------------------------------------------------------------
// helpers
// ---------------------------------------------------------------------------
__device__ __forceinline__ float sigmoidf_(float x) {
    return 1.0f / (1.0f + __expf(-x));
}

// A fragment (16x32 f16) at (row0, k0) of row-major [M][ld] f16.
// Lane l: m = row0 + (l&15), khalf = l>>4.
// ISA 7.12.2: VGPR0-3 = K[8*kh .. +7], VGPR4-7 = K[16+8*kh .. +7].
__device__ __forceinline__ v16h load_a(const _Float16* base, int ld,
                                       int row0, int k0, int lane) {
    const int nl = lane & 15, kh = lane >> 4;
    const _Float16* p = base + (size_t)(row0 + nl) * ld + k0 + kh * 8;
    v16h a;
    ((int4*)&a)[0] = *(const int4*)p;
    ((int4*)&a)[1] = *(const int4*)(p + 16);
    return a;
}

// B fragment (32x16 f16).  B is always W^T with W row-major [N][K], so
// B column n == W row n (contiguous).  Lane l: n = l&15 col, kh = l>>4,
// holds K[8*kh .. +7] and K[16+8*kh .. +7] (same order as A).
__device__ __forceinline__ v16h load_b(const _Float16* W, int ldk,
                                       int n, int k0, int lane) {
    const int kh = lane >> 4;
    const _Float16* p = W + (size_t)n * ldk + k0 + kh * 8;
    v16h b;
    ((int4*)&b)[0] = *(const int4*)p;
    ((int4*)&b)[1] = *(const int4*)(p + 16);
    return b;
}

// ---------------------------------------------------------------------------
// prep kernels
// ---------------------------------------------------------------------------
__global__ void cvt_f16_kernel(const float* __restrict__ src,
                               _Float16* __restrict__ dst, int n) {
    int i = blockIdx.x * blockDim.x + threadIdx.x;
    int stride = gridDim.x * blockDim.x;
    for (; i < n; i += stride) dst[i] = (_Float16)src[i];
}

__global__ void bias_sum_kernel(const float* __restrict__ a,
                                const float* __restrict__ b,
                                float* __restrict__ o, int n) {
    int i = blockIdx.x * blockDim.x + threadIdx.x;
    if (i < n) o[i] = a[i] + b[i];
}

// Gather token embeddings to time-major f16: X[(s*BATCH + b)*HID + e].
__global__ void embed_gather_kernel(const float* __restrict__ E,
                                    const int* __restrict__ tok,
                                    _Float16* __restrict__ X, int Ltok) {
    const int r = blockIdx.x;          // r = s*32 + b
    const int s = r >> 5, b = r & 31;
    const int t = tok[b * Ltok + s];
    X[(size_t)r * HID + threadIdx.x] =
        (_Float16)E[(size_t)t * HID + threadIdx.x];
}

// out[:, 0, :] = 0
__global__ void zero_t0_kernel(float* __restrict__ out) {
    const size_t n = (size_t)BATCH * VOCAB;
    size_t i = (size_t)blockIdx.x * blockDim.x + threadIdx.x;
    const size_t stride = (size_t)gridDim.x * blockDim.x;
    for (; i < n; i += stride) {
        size_t b = i / VOCAB, v = i - b * VOCAB;
        out[b * (size_t)TRGL * VOCAB + v] = 0.0f;
    }
}

// ---------------------------------------------------------------------------
// G^T = (X @ W^T + bias)^T   stored TRANSPOSED: G[n][m], n<1024, m<Mtot.
// X: [Mtot][256] f16, W: [1024][256] f16, bias: [1024] f32.
// Block = 256 thr (8 waves), covers 32 rows (2 m-tiles). Wave w owns
// n-tiles w*128 + j*16, j=0..7.  acc index = j*2 + mt.
// ---------------------------------------------------------------------------
__global__ __launch_bounds__(256)
void gemm_xw_kernel(const _Float16* __restrict__ X,
                    const _Float16* __restrict__ W,
                    const float* __restrict__ bias,
                    float* __restrict__ G, int Mtot) {
    const int mb   = blockIdx.x * 32;        // row base (2 m-tiles)
    const int tid  = threadIdx.x;
    const int lane = tid & 31;
    const int w    = tid >> 5;
    const int nl   = lane & 15, kh = lane >> 4;

    v8f acc[16];
#pragma unroll
    for (int u = 0; u < 16; ++u) acc[u] = (v8f)(0.0f);

    for (int kt = 0; kt < 8; ++kt) {
        const int k0 = kt * 32;
        const v16h a0 = load_a(X, HID, mb,      k0, lane);
        const v16h a1 = load_a(X, HID, mb + 16, k0, lane);
#pragma unroll
        for (int half = 0; half < 2; ++half) {
            v16h b[4];
#pragma unroll
            for (int q = 0; q < 4; ++q) {
                const int n = w * 128 + (half * 4 + q) * 16 + nl;
                b[q] = load_b(W, HID, n, k0, lane);
            }
#pragma unroll
            for (int q = 0; q < 4; ++q) {
                const int j = half * 4 + q;
                acc[j * 2 + 0] = __builtin_amdgcn_wmma_f32_16x16x32_f16(
                    false, a0, false, b[q], (short)0, acc[j * 2 + 0], false, false);
                acc[j * 2 + 1] = __builtin_amdgcn_wmma_f32_16x16x32_f16(
                    false, a1, false, b[q], (short)0, acc[j * 2 + 1], false, false);
            }
        }
    }
    // transposed store: lane owns (n, m=mb+mt*16+kh*8 .. +7) -> contiguous
#pragma unroll
    for (int j = 0; j < 8; ++j) {
        const int n = w * 128 + j * 16 + nl;
        const float bs = bias[n];
#pragma unroll
        for (int mt = 0; mt < 2; ++mt) {
            v8f t = acc[j * 2 + mt] + bs;
            float* dst = G + (size_t)n * Mtot + mb + mt * 16 + kh * 8;
            ((float4*)dst)[0] = ((float4*)&t)[0];
            ((float4*)dst)[1] = ((float4*)&t)[1];
        }
    }
}

// ---------------------------------------------------------------------------
// LSTM recurrence, one persistent workgroup, state in LDS.
// GxT : [1024][Mtot] f32 transposed (x@Wih^T + biases), Mtot = steps*32
// Whh : [1024][256] f16;  h0/c0: [32][256] f32 or nullptr
// Hout: [steps*32][256] f16 (time-major);  hT/cT: [32][256] f32 (optional)
// Block = 256 thr = 8 waves.  Wave w owns hidden cols [32w, 32w+32).
// acc index u = (mt*4 + g)*2 + nt.
// ---------------------------------------------------------------------------
__global__ __launch_bounds__(256)
void lstm_rec_kernel(const float* __restrict__ GxT,
                     const _Float16* __restrict__ Whh,
                     const float* __restrict__ h0,
                     const float* __restrict__ c0,
                     _Float16* __restrict__ Hout,
                     float* __restrict__ hT,
                     float* __restrict__ cT,
                     int steps, int Mtot) {
    __shared__ _Float16 h_sh[BATCH][HID];   // 16 KB
    __shared__ float    c_sh[BATCH][HID];   // 32 KB

    const int tid  = threadIdx.x;
    const int lane = tid & 31;
    const int w    = tid >> 5;
    const int nl   = lane & 15, kh = lane >> 4;

    for (int idx = tid; idx < BATCH * HID; idx += 256) {
        const int m = idx >> 8, j = idx & 255;
        h_sh[m][j] = (_Float16)(h0 ? h0[idx] : 0.0f);
        c_sh[m][j] = c0 ? c0[idx] : 0.0f;
    }
    __syncthreads();

    for (int t = 0; t < steps; ++t) {
        // ---- init accumulators from transposed Gx: 2x b128 per tile ----
        v8f acc[16];
#pragma unroll
        for (int mt = 0; mt < 2; ++mt)
#pragma unroll
            for (int g = 0; g < 4; ++g)
#pragma unroll
                for (int nt = 0; nt < 2; ++nt) {
                    const int n = g * 256 + w * 32 + nt * 16 + nl;
                    const float* p =
                        GxT + (size_t)n * Mtot + t * 32 + mt * 16 + kh * 8;
                    v8f c;
                    ((float4*)&c)[0] = ((const float4*)p)[0];
                    ((float4*)&c)[1] = ((const float4*)p)[1];
                    if (t + 1 < steps)          // next timestep's Gx slice
                        __builtin_prefetch(p + 32, 0, 1);
                    acc[(mt * 4 + g) * 2 + nt] = c;
                }

        // ---- h_{t-1} @ Whh^T via WMMA, K = 256 ----
        for (int kt = 0; kt < 8; ++kt) {
            const int k0 = kt * 32;
            v16h a0, a1;
            {
                const _Float16* p0 = &h_sh[nl][k0 + kh * 8];
                ((int4*)&a0)[0] = *(const int4*)p0;
                ((int4*)&a0)[1] = *(const int4*)(p0 + 16);
                const _Float16* p1 = &h_sh[16 + nl][k0 + kh * 8];
                ((int4*)&a1)[0] = *(const int4*)p1;
                ((int4*)&a1)[1] = *(const int4*)(p1 + 16);
            }
#pragma unroll
            for (int half = 0; half < 2; ++half) {
                v16h b[4];
#pragma unroll
                for (int q = 0; q < 4; ++q) {   // q = gg*2 + nt
                    const int g  = half * 2 + (q >> 1);
                    const int nt = q & 1;
                    const int n  = g * 256 + w * 32 + nt * 16 + nl;
                    b[q] = load_b(Whh, HID, n, k0, lane);
                }
#pragma unroll
                for (int q = 0; q < 4; ++q) {
                    const int g  = half * 2 + (q >> 1);
                    const int nt = q & 1;
                    const int u0 = (0 * 4 + g) * 2 + nt;
                    const int u1 = (1 * 4 + g) * 2 + nt;
                    acc[u0] = __builtin_amdgcn_wmma_f32_16x16x32_f16(
                        false, a0, false, b[q], (short)0, acc[u0], false, false);
                    acc[u1] = __builtin_amdgcn_wmma_f32_16x16x32_f16(
                        false, a1, false, b[q], (short)0, acc[u1], false, false);
                }
            }
        }
        __syncthreads();   // all reads of h_sh complete before update

        // ---- elementwise cell update (gate order: i, f, g, o) ----
#pragma unroll
        for (int mt = 0; mt < 2; ++mt)
#pragma unroll
            for (int nt = 0; nt < 2; ++nt) {
#pragma unroll
                for (int i = 0; i < 8; ++i) {
                    const int m = mt * 16 + kh * 8 + i;
                    const int j = w * 32 + nt * 16 + nl;
                    const float gi = acc[(mt * 4 + 0) * 2 + nt][i];
                    const float gf = acc[(mt * 4 + 1) * 2 + nt][i];
                    const float gg = acc[(mt * 4 + 2) * 2 + nt][i];
                    const float go = acc[(mt * 4 + 3) * 2 + nt][i];
                    const float cn = sigmoidf_(gf) * c_sh[m][j] +
                                     sigmoidf_(gi) * tanhf(gg);
                    const float hn = sigmoidf_(go) * tanhf(cn);
                    c_sh[m][j] = cn;
                    h_sh[m][j] = (_Float16)hn;
                }
            }
        __syncthreads();   // h/c visible to all waves

        // ---- bulk h_t -> Hout copy, b128 both sides ----
        {
            const int4* s4 = (const int4*)&h_sh[0][0];
            int4* d4 = (int4*)(Hout + (size_t)t * BATCH * HID);
            for (int q = tid; q < BATCH * HID * 2 / 16; q += 256)
                d4[q] = s4[q];   // values land in regs before next overwrite
        }
    }

    if (hT != nullptr) {
        for (int idx = tid; idx < BATCH * HID; idx += 256) {
            const int m = idx >> 8, j = idx & 255;
            hT[idx] = (float)h_sh[m][j];
            cT[idx] = c_sh[m][j];
        }
    }
}

// ---------------------------------------------------------------------------
// logits = Hdec @ fc_w^T + fc_b -> out[B][T][V] at t+1.
// Hdec: [2016][256] f16 (rows r = t*32 + b), Wf: [V][256] f16.
// grid = (21 blocks of 6 m-tiles, 197 col chunks of 256); 8 waves/block;
// wave w owns n-tiles cb*256 + w*32 + {0,16}. acc = [mm][nt] -> mm*2+nt.
// Each B fragment feeds 6 WMMAs (fc_w L2 traffic cut 6x).
// ---------------------------------------------------------------------------
__global__ __launch_bounds__(256)
void gemm_fc_kernel(const _Float16* __restrict__ Hdec,
                    const _Float16* __restrict__ Wf,
                    const float* __restrict__ bf,
                    float* __restrict__ out) {
    const int mtb  = blockIdx.x * 6;
    const int cb   = blockIdx.y;
    const int tid  = threadIdx.x;
    const int lane = tid & 31;
    const int w    = tid >> 5;
    const int nl   = lane & 15, kh = lane >> 4;

    int ncl[2];
#pragma unroll
    for (int nt = 0; nt < 2; ++nt) {
        int n = cb * 256 + w * 32 + nt * 16 + nl;
        ncl[nt] = (n >= VOCAB) ? (VOCAB - 1) : n;   // clamp loads
    }

    v8f acc[12];
#pragma unroll
    for (int u = 0; u < 12; ++u) acc[u] = (v8f)(0.0f);

    for (int kt = 0; kt < 8; ++kt) {
        const int k0 = kt * 32;
        v16h a[6];
#pragma unroll
        for (int mm = 0; mm < 6; ++mm)
            a[mm] = load_a(Hdec, HID, (mtb + mm) * 16, k0, lane);
        const v16h b0 = load_b(Wf, HID, ncl[0], k0, lane);
        const v16h b1 = load_b(Wf, HID, ncl[1], k0, lane);
#pragma unroll
        for (int mm = 0; mm < 6; ++mm)
            acc[mm * 2 + 0] = __builtin_amdgcn_wmma_f32_16x16x32_f16(
                false, a[mm], false, b0, (short)0, acc[mm * 2 + 0], false, false);
#pragma unroll
        for (int mm = 0; mm < 6; ++mm)
            acc[mm * 2 + 1] = __builtin_amdgcn_wmma_f32_16x16x32_f16(
                false, a[mm], false, b1, (short)0, acc[mm * 2 + 1], false, false);
    }
#pragma unroll
    for (int nt = 0; nt < 2; ++nt) {
        const int n = cb * 256 + w * 32 + nt * 16 + nl;
        if (n < VOCAB) {
            const float bs = bf[n];
#pragma unroll
            for (int mm = 0; mm < 6; ++mm) {
#pragma unroll
                for (int i = 0; i < 8; ++i) {
                    const int m = (mtb + mm) * 16 + kh * 8 + i;  // < 2016
                    const int t = m >> 5, b = m & 31;
                    out[(size_t)b * TRGL * VOCAB + (size_t)(t + 1) * VOCAB + n] =
                        acc[mm * 2 + nt][i] + bs;
                }
            }
        }
    }
}

// ---------------------------------------------------------------------------
// host side
// ---------------------------------------------------------------------------
extern "C" void kernel_launch(void* const* d_in, const int* in_sizes, int n_in,
                              void* d_out, int out_size, void* d_ws,
                              size_t ws_size, hipStream_t stream) {
    (void)in_sizes; (void)n_in; (void)out_size; (void)ws_size;

    // setup_inputs() insertion order, nested structures flattened in order:
    const int*   src     = (const int*)d_in[0];              // [32][96]
    const int*   trg     = (const int*)d_in[1];              // [32][64]
    const float* embed_w = (const float*)d_in[2];            // [V][256]
    const float* wih[4], *whh[4], *bih[4], *bhh[4];
    for (int l = 0; l < 4; ++l) {                            // enc0,enc1,dec0,dec1
        wih[l] = (const float*)d_in[3 + 4 * l + 0];          // [1024][256]
        whh[l] = (const float*)d_in[3 + 4 * l + 1];          // [1024][256]
        bih[l] = (const float*)d_in[3 + 4 * l + 2];          // [1024]
        bhh[l] = (const float*)d_in[3 + 4 * l + 3];          // [1024]
    }
    const float* fc_w = (const float*)d_in[19];              // [V][256]
    const float* fc_b = (const float*)d_in[20];              // [V]
    float* out = (float*)d_out;                              // [32][64][V]

    // workspace carve-up (256 B aligned)
    size_t off = 0;
    char* base = (char*)d_ws;
    auto carve = [&](size_t bytes) {
        off = (off + 255) & ~(size_t)255;
        void* p = base + off;
        off += bytes;
        return p;
    };
    _Float16* Xe     = (_Float16*)carve((size_t)SRCL * BATCH * HID * 2);
    _Float16* Xd     = (_Float16*)carve((size_t)DSTEP * BATCH * HID * 2);
    _Float16* wih_h[4], *whh_h[4];
    float*    bsum[4];
    for (int l = 0; l < 4; ++l) {
        wih_h[l] = (_Float16*)carve((size_t)G4H * HID * 2);
        whh_h[l] = (_Float16*)carve((size_t)G4H * HID * 2);
        bsum[l]  = (float*)carve((size_t)G4H * 4);
    }
    _Float16* fcw_h  = (_Float16*)carve((size_t)VOCAB * HID * 2);
    float*    gx_enc = (float*)carve((size_t)SRCL * BATCH * G4H * 4);  // [1024][3072]
    float*    gx_dec = (float*)carve((size_t)DSTEP * BATCH * G4H * 4); // [1024][2016]
    _Float16* H1e    = (_Float16*)carve((size_t)SRCL * BATCH * HID * 2);
    _Float16* H2e    = (_Float16*)carve((size_t)SRCL * BATCH * HID * 2);
    _Float16* H1d    = (_Float16*)carve((size_t)DSTEP * BATCH * HID * 2);
    _Float16* H2d    = (_Float16*)carve((size_t)DSTEP * BATCH * HID * 2);
    float* hT_e1 = (float*)carve((size_t)BATCH * HID * 4);
    float* cT_e1 = (float*)carve((size_t)BATCH * HID * 4);
    float* hT_e2 = (float*)carve((size_t)BATCH * HID * 4);
    float* cT_e2 = (float*)carve((size_t)BATCH * HID * 4);

    const int ME = SRCL * BATCH;    // 3072
    const int MD = DSTEP * BATCH;   // 2016

    // ---- prep: f32->f16 weights, fused biases, embedding gathers ----
    for (int l = 0; l < 4; ++l) {
        cvt_f16_kernel<<<1024, 256, 0, stream>>>(wih[l], wih_h[l], G4H * HID);
        cvt_f16_kernel<<<1024, 256, 0, stream>>>(whh[l], whh_h[l], G4H * HID);
        bias_sum_kernel<<<4, 256, 0, stream>>>(bih[l], bhh[l], bsum[l], G4H);
    }
    cvt_f16_kernel<<<8192, 256, 0, stream>>>(fc_w, fcw_h, VOCAB * HID);
    embed_gather_kernel<<<ME, HID, 0, stream>>>(embed_w, src, Xe, SRCL);
    embed_gather_kernel<<<MD, HID, 0, stream>>>(embed_w, trg, Xd, TRGL);

    // ---- encoder ----
    gemm_xw_kernel<<<ME / 32, 256, 0, stream>>>(Xe, wih_h[0], bsum[0], gx_enc, ME);
    lstm_rec_kernel<<<1, 256, 0, stream>>>(gx_enc, whh_h[0], nullptr, nullptr,
                                           H1e, hT_e1, cT_e1, SRCL, ME);
    gemm_xw_kernel<<<ME / 32, 256, 0, stream>>>(H1e, wih_h[1], bsum[1], gx_enc, ME);
    lstm_rec_kernel<<<1, 256, 0, stream>>>(gx_enc, whh_h[1], nullptr, nullptr,
                                           H2e, hT_e2, cT_e2, SRCL, ME);

    // ---- decoder (teacher forcing, init = encoder final state) ----
    gemm_xw_kernel<<<MD / 32, 256, 0, stream>>>(Xd, wih_h[2], bsum[2], gx_dec, MD);
    lstm_rec_kernel<<<1, 256, 0, stream>>>(gx_dec, whh_h[2], hT_e1, cT_e1,
                                           H1d, nullptr, nullptr, DSTEP, MD);
    gemm_xw_kernel<<<MD / 32, 256, 0, stream>>>(H1d, wih_h[3], bsum[3], gx_dec, MD);
    lstm_rec_kernel<<<1, 256, 0, stream>>>(gx_dec, whh_h[3], hT_e2, cT_e2,
                                           H2d, nullptr, nullptr, DSTEP, MD);

    // ---- output: zero t=0 slice, then FC logits into t=1..63 ----
    zero_t0_kernel<<<2048, 256, 0, stream>>>(out);
    dim3 fcGrid((MD / 16) / 6, (VOCAB + 255) / 256);   // (21, 197)
    gemm_fc_kernel<<<fcGrid, 256, 0, stream>>>(H2d, fcw_h, fc_b, out);
}
